// RegionNonLocalEnhancedDenseBlock_17703855194548
// MI455X (gfx1250) — compile-verified, matrix operand order checked by reading, not run on hardware
//
#include <hip/hip_runtime.h>
#include <hip/hip_bf16.h>

typedef __bf16 bf16_t;
typedef __attribute__((ext_vector_type(16))) __bf16 v16bf;
typedef __attribute__((ext_vector_type(8)))  float  v8f;

#define HDIM 256
#define WDIM 256
#define CIN  64
#define CLAT 32

// ---------------------------------------------------------------------------
// WMMA helpers (CDNA5 16x16x32 bf16, f32 accumulate)
// ---------------------------------------------------------------------------
__device__ __forceinline__ v8f wmma_bf(v16bf a, v16bf b, v8f c) {
  return __builtin_amdgcn_wmma_f32_16x16x32_bf16(false, a, false, b,
                                                 (short)0, c, false, false);
}

__device__ __forceinline__ v8f vzero() {
  v8f z = {0.f, 0.f, 0.f, 0.f, 0.f, 0.f, 0.f, 0.f};
  return z;
}

// K-pair base within a K=32 slab for bf16 A/B fragments (ISA 7.12.2):
// lanes 0-15: VGPR i<4 -> K=2i,2i+1 ; i>=4 -> K=16+2(i-4) ; lanes 16-31: +8
__device__ __forceinline__ int wk_base(int i, int half) {
  return ((i < 4) ? (2 * i) : (16 + 2 * (i - 4))) + 8 * half;
}

// A/B fragment from bf16 matrix, lane-dim slow (row-major): elem(r,k)=base[r*ldm+k]
__device__ __forceinline__ v16bf frag_ld_bf(const bf16_t* __restrict__ base,
                                            int ldm, int i0, int koff) {
  int lane = threadIdx.x & 31;
  int r = i0 + (lane & 15);
  int half = lane >> 4;
  v16bf f;
#pragma unroll
  for (int i = 0; i < 8; ++i) {
    int k = koff + wk_base(i, half);
    f[2 * i]     = base[r * ldm + k];
    f[2 * i + 1] = base[r * ldm + k + 1];
  }
  return f;
}

// fragment with K slow dim: elem(k, n)=base[k*ldm + n]
__device__ __forceinline__ v16bf frag_ld_bf_T(const bf16_t* __restrict__ base,
                                              int ldm, int i0, int koff) {
  int lane = threadIdx.x & 31;
  int n = i0 + (lane & 15);
  int half = lane >> 4;
  v16bf f;
#pragma unroll
  for (int i = 0; i < 8; ++i) {
    int k = koff + wk_base(i, half);
    f[2 * i]     = base[k * ldm + n];
    f[2 * i + 1] = base[(k + 1) * ldm + n];
  }
  return f;
}

// fragment from f32 matrix (row-major, lane-dim slow), convert to bf16
__device__ __forceinline__ v16bf frag_ld_f32(const float* __restrict__ base,
                                             int ldm, int i0, int koff) {
  int lane = threadIdx.x & 31;
  int r = i0 + (lane & 15);
  int half = lane >> 4;
  v16bf f;
#pragma unroll
  for (int i = 0; i < 8; ++i) {
    int k = koff + wk_base(i, half);
    f[2 * i]     = (bf16_t)base[r * ldm + k];
    f[2 * i + 1] = (bf16_t)base[r * ldm + k + 1];
  }
  return f;
}

// fragment from f32 matrix with K slow (stride ldm, e.g. NCHW channel stride)
__device__ __forceinline__ v16bf frag_ld_f32_T(const float* __restrict__ base,
                                               size_t ldm, int i0, int koff) {
  int lane = threadIdx.x & 31;
  int n = i0 + (lane & 15);
  int half = lane >> 4;
  v16bf f;
#pragma unroll
  for (int i = 0; i < 8; ++i) {
    int k = koff + wk_base(i, half);
    f[2 * i]     = (bf16_t)base[(size_t)k * ldm + n];
    f[2 * i + 1] = (bf16_t)base[(size_t)(k + 1) * ldm + n];
  }
  return f;
}

// ---------------------------------------------------------------------------
// Kernel 0: convert all GEMM weights to bf16 in workspace.
// Conv weight [O][I][3][3] -> [O][I*9] is an identity reindex, so pure convert.
// ---------------------------------------------------------------------------
__global__ void prep_weights(const float* __restrict__ W1, const float* __restrict__ W2,
                             const float* __restrict__ W3, const float* __restrict__ Wf,
                             const float* __restrict__ Wo, const float* __restrict__ Wq,
                             const float* __restrict__ Wk, const float* __restrict__ Wv,
                             bf16_t* __restrict__ o1, bf16_t* __restrict__ o2,
                             bf16_t* __restrict__ o3, bf16_t* __restrict__ of,
                             bf16_t* __restrict__ oo, bf16_t* __restrict__ oq,
                             bf16_t* __restrict__ ok, bf16_t* __restrict__ ov) {
  int i = blockIdx.x * blockDim.x + threadIdx.x;
  if (i < 32 * 576)  o1[i] = (bf16_t)W1[i];
  if (i < 32 * 864)  o2[i] = (bf16_t)W2[i];
  if (i < 32 * 1152) o3[i] = (bf16_t)W3[i];
  if (i < 64 * 160)  of[i] = (bf16_t)Wf[i];
  if (i < 64 * 32)   oo[i] = (bf16_t)Wo[i];
  if (i < 32 * 64) {
    oq[i] = (bf16_t)Wq[i];
    ok[i] = (bf16_t)Wk[i];
    ov[i] = (bf16_t)Wv[i];
  }
}

// ---------------------------------------------------------------------------
// Kernel 1: region non-local attention. One block per 32x32 cell (256 blocks).
// All GEMMs (k/v proj+pool, q proj, QK^T, PV, Wo proj) on the WMMA pipe.
// Writes f0 (bf16) into channels 0..63 of the concat buffer [4][160][256][256].
// ---------------------------------------------------------------------------
__global__ __launch_bounds__(256) void attn_kernel(
    const float* __restrict__ x,
    const bf16_t* __restrict__ Wq_bf, const float* __restrict__ bq,
    const bf16_t* __restrict__ Wk_bf, const float* __restrict__ bk,
    const bf16_t* __restrict__ Wv_bf, const float* __restrict__ bv,
    const float* __restrict__ gamma_p, const float* __restrict__ bo,
    const bf16_t* __restrict__ Wo_bf, bf16_t* __restrict__ cat) {
  __shared__ bf16_t Kp[256 * 32];   // pooled keys   [key][lat]
  __shared__ bf16_t Vp[256 * 32];   // pooled values [key][lat]
  __shared__ bf16_t ql[128 * 32];   // query chunk   [q][lat]
  __shared__ float  scratchf[4096]; // phase A: k/v pre-pool; phase B: per-wave staging

  const int bid = blockIdx.x;
  const int n = bid >> 6, gh = (bid >> 3) & 7, gw = bid & 7;
  const int row0 = gh * 32, col0 = gw * 32;
  const int t = threadIdx.x, lane = t & 31, wave = t >> 5;
  const int half = lane >> 4, coln = lane & 15;
  const float gamma = gamma_p[0];
  const size_t HW = (size_t)HDIM * WDIM;
  const float* xcell = x + (size_t)n * CIN * HW;  // + yy*WDIM + xx per use

  // ---- Phase A (WMMA): k/v projection + 2x2 maxpool, one row-pair at a time.
  // Waves 0-3 compute k, waves 4-7 compute v; wave owns one 16-px tile of the
  // 64 px (2 rows x 32 cols) in the row-pair. Then all threads pool.
  {
    const int matsel = wave >> 2;  // 0 = k, 1 = v
    const int pt = wave & 3;       // px tile: (row pt>>1, cols (pt&1)*16..)
    const bf16_t* Wm = matsel ? Wv_bf : Wk_bf;
    const float* bm = matsel ? bv : bk;
    const v16bf a00 = frag_ld_bf(Wm, 64, 0, 0);
    const v16bf a01 = frag_ld_bf(Wm, 64, 0, 32);
    const v16bf a10 = frag_ld_bf(Wm, 64, 16, 0);
    const v16bf a11 = frag_ld_bf(Wm, 64, 16, 32);
    float* stg = scratchf + matsel * 2048;  // [64 px][32 ch] f32
    for (int rp = 0; rp < 16; ++rp) {
      const int yy = row0 + rp * 2 + (pt >> 1);
      const int xb = (pt & 1) * 16;
      const float* xp = xcell + (size_t)yy * WDIM + col0 + xb;
      v16bf b0 = frag_ld_f32_T(xp, HW, 0, 0);   // B: lane=px, K=in-ch 0..31
      v16bf b1 = frag_ld_f32_T(xp, HW, 0, 32);  //                     32..63
      v8f c0 = vzero(), c1 = vzero();
      c0 = wmma_bf(a00, b0, c0);
      c0 = wmma_bf(a01, b1, c0);
      c1 = wmma_bf(a10, b0, c1);
      c1 = wmma_bf(a11, b1, c1);
      const int pl = pt * 16 + coln;  // local px 0..63 (= row*32 + col)
#pragma unroll
      for (int r = 0; r < 8; ++r) {
        const int c = r + 8 * half;
        stg[pl * 32 + c]      = c0[r] + bm[c];
        stg[pl * 32 + 16 + c] = c1[r] + bm[16 + c];
      }
      __syncthreads();
      // 2x2 maxpool: 16 pooled cols x 32 ch per matrix; 2 outputs per thread
#pragma unroll
      for (int j = 0; j < 2; ++j) {
        const int idx = t * 2 + j;  // 0..511
        const int pc = idx >> 5, ch = idx & 31;
        const int p00 = (2 * pc) * 32 + ch, p01 = (2 * pc + 1) * 32 + ch;
        const int p10 = (32 + 2 * pc) * 32 + ch, p11 = (32 + 2 * pc + 1) * 32 + ch;
        // stage layout: [pl][ch] with pl = row*32 + col  ->  index pl*32+ch
        const float km = fmaxf(fmaxf(scratchf[(2 * pc) * 32 + ch],
                                     scratchf[(2 * pc + 1) * 32 + ch]),
                               fmaxf(scratchf[(32 + 2 * pc) * 32 + ch],
                                     scratchf[(32 + 2 * pc + 1) * 32 + ch]));
        const float vm = fmaxf(fmaxf(scratchf[2048 + (2 * pc) * 32 + ch],
                                     scratchf[2048 + (2 * pc + 1) * 32 + ch]),
                               fmaxf(scratchf[2048 + (32 + 2 * pc) * 32 + ch],
                                     scratchf[2048 + (32 + 2 * pc + 1) * 32 + ch]));
        Kp[(rp * 16 + pc) * 32 + ch] = (bf16_t)km;
        Vp[(rp * 16 + pc) * 32 + ch] = (bf16_t)vm;
        (void)p00; (void)p01; (void)p10; (void)p11;
      }
      __syncthreads();
    }
  }

  // ---- Phase B: 8 chunks of 128 queries; each wave owns one 16-query tile ----
  for (int chunk = 0; chunk < 8; ++chunk) {
    {  // q projection (WMMA): wave w -> queries [w*16, w*16+16) of the chunk
      const int yy = row0 + chunk * 4 + (wave >> 1);
      const int xb = (wave & 1) * 16;
      const float* xp = xcell + (size_t)yy * WDIM + col0 + xb;
      v16bf b0 = frag_ld_f32_T(xp, HW, 0, 0);
      v16bf b1 = frag_ld_f32_T(xp, HW, 0, 32);
      v8f c0 = vzero(), c1 = vzero();
      c0 = wmma_bf(frag_ld_bf(Wq_bf, 64, 0, 0), b0, c0);
      c0 = wmma_bf(frag_ld_bf(Wq_bf, 64, 0, 32), b1, c0);
      c1 = wmma_bf(frag_ld_bf(Wq_bf, 64, 16, 0), b0, c1);
      c1 = wmma_bf(frag_ld_bf(Wq_bf, 64, 16, 32), b1, c1);
#pragma unroll
      for (int r = 0; r < 8; ++r) {
        const int c = r + 8 * half;
        ql[(wave * 16 + coln) * 32 + c]      = (bf16_t)(c0[r] + bq[c]);
        ql[(wave * 16 + coln) * 32 + 16 + c] = (bf16_t)(c1[r] + bq[16 + c]);
      }
    }
    __syncthreads();

    const int qtb = wave * 16;  // query tile base within chunk
    // scores: one WMMA per 16-key tile (K = 32 latent channels)
    v16bf aq = frag_ld_bf(ql, 32, qtb, 0);
    v8f sc[16];
#pragma unroll
    for (int kt = 0; kt < 16; ++kt) {
      v16bf kbf = frag_ld_bf(Kp, 32, kt * 16, 0);  // B: lane=key, K=lat
      sc[kt] = wmma_bf(aq, kbf, vzero());
    }
    // softmax over 256 keys; rows live as (vgpr r, lane-half), cols across lanes
    float rmax[8], rinv[8];
#pragma unroll
    for (int r = 0; r < 8; ++r) {
      float m = -3.4e38f;
#pragma unroll
      for (int kt = 0; kt < 16; ++kt) m = fmaxf(m, sc[kt][r]);
      for (int sh = 1; sh < 16; sh <<= 1) m = fmaxf(m, __shfl_xor(m, sh, 16));
      rmax[r] = m;
    }
#pragma unroll
    for (int r = 0; r < 8; ++r) {
      float s = 0.f;
#pragma unroll
      for (int kt = 0; kt < 16; ++kt) {
        float e = __expf(sc[kt][r] - rmax[r]);
        sc[kt][r] = e;
        s += e;
      }
      for (int sh = 1; sh < 16; sh <<= 1) s += __shfl_xor(s, sh, 16);
      rinv[r] = 1.f / s;
    }
#pragma unroll
    for (int r = 0; r < 8; ++r)
#pragma unroll
      for (int kt = 0; kt < 16; ++kt) sc[kt][r] *= rinv[r];

    // out[16q][32lat] = P @ V^T, staged through per-wave LDS for A layout
    v8f o0 = vzero(), o1 = vzero();
    float* st = scratchf + wave * 512;
#pragma unroll
    for (int ks = 0; ks < 8; ++ks) {
#pragma unroll
      for (int r = 0; r < 8; ++r) {
        st[(r + 8 * half) * 32 + coln]      = sc[2 * ks][r];
        st[(r + 8 * half) * 32 + 16 + coln] = sc[2 * ks + 1][r];
      }
      v16bf pa  = frag_ld_f32(st, 32, 0, 0);          // A: probs [16][32]
      v16bf vb0 = frag_ld_bf_T(Vp, 32, 0, ks * 32);   // B: V^T, lat 0-15
      v16bf vb1 = frag_ld_bf_T(Vp, 32, 16, ks * 32);  // B: V^T, lat 16-31
      o0 = wmma_bf(pa, vb0, o0);
      o1 = wmma_bf(pa, vb1, o1);
    }

    // f0[64][16q] = gamma*(Wo @ out^T + bo) + x ; out^T staged for B fragment
#pragma unroll
    for (int r = 0; r < 8; ++r) {
      st[(r + 8 * half) * 32 + coln]      = o0[r];
      st[(r + 8 * half) * 32 + 16 + coln] = o1[r];
    }
    v16bf pb = frag_ld_f32(st, 32, 0, 0);  // B: lane=q, K=lat (transposed read)
#pragma unroll
    for (int mt = 0; mt < 4; ++mt) {
      v16bf wa = frag_ld_bf(Wo_bf, 32, mt * 16, 0);
      v8f f0 = wmma_bf(wa, pb, vzero());
      const int qi = chunk * 128 + qtb + coln;
      const int yy = row0 + (qi >> 5), xx = col0 + (qi & 31);
#pragma unroll
      for (int r = 0; r < 8; ++r) {
        const int oc = mt * 16 + r + 8 * half;
        const float val =
            gamma * (f0[r] + bo[oc]) +
            x[(((size_t)n * CIN + oc) * HDIM + yy) * WDIM + xx];
        cat[(((size_t)n * 160 + oc) * HDIM + yy) * WDIM + xx] = (bf16_t)val;
      }
    }
    __syncthreads();  // ql / scratchf reused next chunk
  }
}

// ---------------------------------------------------------------------------
// Kernels 2-4: implicit-GEMM 3x3 conv, pad=1, relu. Input = first KTOT/9
// channels of concat buffer; output -> channels [ocBase, ocBase+32).
// Block tile: 32 oc x 128 px (one row segment); wave w owns x-subtile w.
// ---------------------------------------------------------------------------
template <int KTOT>
__global__ __launch_bounds__(256) void conv3x3_kernel(
    const bf16_t* __restrict__ cat, const bf16_t* __restrict__ Wr,
    const float* __restrict__ bias, bf16_t* __restrict__ out_cat, int ocBase) {
  const int bid = blockIdx.x;  // 4 imgs * 256 rows * 2 xsegs
  const int n = bid >> 9, rem = bid & 511;
  const int y = rem >> 1, xseg = rem & 1;
  const int t = threadIdx.x, lane = t & 31, wave = t >> 5;
  const int half = lane >> 4, coln = lane & 15;
  const int xx = xseg * 128 + wave * 16 + coln;
  const size_t HW = (size_t)HDIM * WDIM;
  const bf16_t* inb = cat + (size_t)n * 160 * HW;

  v8f a0 = vzero(), a1 = vzero();
  constexpr int KS = KTOT / 32;
  for (int ks = 0; ks < KS; ++ks) {
    if (ks + 1 < KS) __builtin_prefetch(Wr + (ks + 1) * 32, 0, 0);
    v16bf wa0 = frag_ld_bf(Wr, KTOT, 0, ks * 32);
    v16bf wa1 = frag_ld_bf(Wr, KTOT, 16, ks * 32);
    v16bf bfrag;  // im2col B fragment: k = ic*9 + ky*3 + kx, n = pixel col
#pragma unroll
    for (int i = 0; i < 8; ++i) {
      const int k0 = ks * 32 + wk_base(i, half);
#pragma unroll
      for (int j = 0; j < 2; ++j) {
        const int k = k0 + j;
        const int ic = k / 9, r9 = k - ic * 9;
        const int dy = r9 / 3 - 1, dx = r9 - (r9 / 3) * 3 - 1;
        const int yy = y + dy, x2 = xx + dx;
        bf16_t v = (bf16_t)0.f;
        if ((unsigned)yy < HDIM && (unsigned)x2 < WDIM)
          v = inb[((size_t)ic * HDIM + yy) * WDIM + x2];
        bfrag[2 * i + j] = v;
      }
    }
    a0 = wmma_bf(wa0, bfrag, a0);
    a1 = wmma_bf(wa1, bfrag, a1);
  }
#pragma unroll
  for (int r = 0; r < 8; ++r) {
    const int oc0 = r + 8 * half, oc1 = 16 + r + 8 * half;
    const float v0 = fmaxf(a0[r] + bias[oc0], 0.f);
    const float v1 = fmaxf(a1[r] + bias[oc1], 0.f);
    out_cat[(((size_t)n * 160 + ocBase + oc0) * HDIM + y) * WDIM + xx] = (bf16_t)v0;
    out_cat[(((size_t)n * 160 + ocBase + oc1) * HDIM + y) * WDIM + xx] = (bf16_t)v1;
  }
}

// ---------------------------------------------------------------------------
// Kernel 5: 1x1 fusion over 160 channels + bias + residual x, fp32 out.
// Block tile: 64 oc x 32 px; 8 waves = 4 oc-tiles x 2 px-tiles.
// ---------------------------------------------------------------------------
__global__ __launch_bounds__(256) void fuse_kernel(
    const bf16_t* __restrict__ cat, const bf16_t* __restrict__ Wf_bf,
    const float* __restrict__ bfus, const float* __restrict__ x,
    float* __restrict__ out) {
  const int bid = blockIdx.x;  // 4 imgs * 256 rows * 8 xsegs
  const int n = bid >> 11, rem = bid & 2047;
  const int y = rem >> 3, xseg = rem & 7;
  const int t = threadIdx.x, lane = t & 31, wave = t >> 5;
  const int mt = wave & 3, nt = wave >> 2;
  const int half = lane >> 4, coln = lane & 15;
  const int xx = xseg * 32 + nt * 16 + coln;
  const size_t HW = (size_t)HDIM * WDIM;
  const bf16_t* inb =
      cat + ((size_t)n * 160 * HDIM + y) * WDIM + xseg * 32 + nt * 16;

  v8f acc = vzero();
#pragma unroll
  for (int ks = 0; ks < 5; ++ks) {
    v16bf wa = frag_ld_bf(Wf_bf, 160, mt * 16, ks * 32);
    v16bf bb = frag_ld_bf_T(inb, (int)HW, 0, ks * 32);  // k = channel (slow)
    acc = wmma_bf(wa, bb, acc);
  }
#pragma unroll
  for (int r = 0; r < 8; ++r) {
    const int oc = mt * 16 + r + 8 * half;
    const size_t xi = (((size_t)n * CIN + oc) * HDIM + y) * WDIM + xx;
    out[xi] = acc[r] + bfus[oc] + x[xi];
  }
}

// ---------------------------------------------------------------------------
extern "C" void kernel_launch(void* const* d_in, const int* in_sizes, int n_in,
                              void* d_out, int out_size, void* d_ws,
                              size_t ws_size, hipStream_t stream) {
  const float* x  = (const float*)d_in[0];
  const float* Wq = (const float*)d_in[1];
  const float* bq = (const float*)d_in[2];
  const float* Wk = (const float*)d_in[3];
  const float* bk = (const float*)d_in[4];
  const float* Wv = (const float*)d_in[5];
  const float* bv = (const float*)d_in[6];
  const float* Wo = (const float*)d_in[7];
  const float* bo = (const float*)d_in[8];
  const float* gm = (const float*)d_in[9];
  const float* W1 = (const float*)d_in[10];
  const float* b1 = (const float*)d_in[11];
  const float* W2 = (const float*)d_in[12];
  const float* b2 = (const float*)d_in[13];
  const float* W3 = (const float*)d_in[14];
  const float* b3 = (const float*)d_in[15];
  const float* Wf = (const float*)d_in[16];
  const float* bfus = (const float*)d_in[17];
  float* out = (float*)d_out;

  // workspace layout: bf16 concat [4][160][256][256] then bf16 weights
  const size_t catElems = (size_t)4 * 160 * HDIM * WDIM;
  bf16_t* cat = (bf16_t*)d_ws;
  bf16_t* W1r = cat + catElems;
  bf16_t* W2r = W1r + 32 * 576;
  bf16_t* W3r = W2r + 32 * 864;
  bf16_t* Wfb = W3r + 32 * 1152;
  bf16_t* Wob = Wfb + 64 * 160;
  bf16_t* Wqb = Wob + 64 * 32;
  bf16_t* Wkb = Wqb + 32 * 64;
  bf16_t* Wvb = Wkb + 32 * 64;

  prep_weights<<<144, 256, 0, stream>>>(W1, W2, W3, Wf, Wo, Wq, Wk, Wv, W1r,
                                        W2r, W3r, Wfb, Wob, Wqb, Wkb, Wvb);
  attn_kernel<<<256, 256, 0, stream>>>(x, Wqb, bq, Wkb, bk, Wvb, bv, gm, bo,
                                       Wob, cat);
  conv3x3_kernel<576><<<2048, 256, 0, stream>>>(cat, W1r, b1, cat, 64);
  conv3x3_kernel<864><<<2048, 256, 0, stream>>>(cat, W2r, b2, cat, 96);
  conv3x3_kernel<1152><<<2048, 256, 0, stream>>>(cat, W3r, b3, cat, 128);
  fuse_kernel<<<8192, 256, 0, stream>>>(cat, Wfb, bfus, x, out);
}
